// CrossAttention_52364241273279
// MI455X (gfx1250) — matrix operand order repaired and checked by reading, not compiled, not run on hardware
//
#include <hip/hip_runtime.h>
#include <hip/hip_bf16.h>
#include <math.h>

// Problem constants (match reference)
#define DIM    1024
#define NBATCH 4
#define SQ     2048
#define SK     2048

// Shared tiling parameters
#define BLK_M  128
#define BLK_K  32
#define A_PITCH 40   // bf16 elems per LDS row (32 + 8 pad) = 80B -> bank spread
#define B_PITCH 40

// proj kernel tile (N per block)
#define PRJ_N  64
// bf16 GEMM kernel tile (N per block)
#define GEM_N  128

typedef __attribute__((ext_vector_type(16))) __bf16 v16bf;
typedef __attribute__((ext_vector_type(8)))  __bf16 v8bf;
typedef __attribute__((ext_vector_type(8)))  float  v8f;
typedef __attribute__((ext_vector_type(4)))  float  v4f;

// ---------------------------------------------------------------------------
// CDNA5 async global->LDS copy path (ASYNCcnt-tracked), with sync fallback.
// ROCm clang signature (from probe diagnostic): arg0 = v4i in AS(1) (printed
// as "__device__"), arg1 = LDS side, then two integer immediates.
// ---------------------------------------------------------------------------
#if defined(__HIP_DEVICE_COMPILE__) &&                                     \
    __has_builtin(__builtin_amdgcn_global_load_async_to_lds_b128) &&       \
    __has_builtin(__builtin_amdgcn_s_wait_asynccnt)
#define USE_ASYNC_COPY 1
#else
#define USE_ASYNC_COPY 0
#endif

#if USE_ASYNC_COPY
typedef int v4i_gcc __attribute__((vector_size(16)));
typedef __attribute__((address_space(1))) v4i_gcc gv4i;
typedef __attribute__((address_space(3))) v4i_gcc lv4i;
__device__ __forceinline__ void async_ld_b128(const __bf16* g, __bf16* l) {
  __builtin_amdgcn_global_load_async_to_lds_b128((gv4i*)g, (lv4i*)l, 0, 0);
}
#endif

__device__ __forceinline__ v8f wmma_bf16f32(v16bf a, v16bf b, v8f c) {
#if defined(__HIP_DEVICE_COMPILE__)
  // 8 args: (neg_a, A, neg_b, B, c_mod, C, reuse_a, reuse_b)
  return __builtin_amdgcn_wmma_f32_16x16x32_bf16(false, a, false, b, (short)0, c,
                                                 false, false);
#else
  (void)a; (void)b;
  return c;
#endif
}

// A fragment, 16x32 bf16 (MxK). ISA layout: lane<16 holds M=lane, K={0..7,16..23};
// lane>=16 holds M=lane-16, K={8..15,24..31}. Two 16B LDS reads.
__device__ __forceinline__ v16bf frag_a(const __bf16* lds, int mBase, int lane) {
  const __bf16* p = lds + (mBase + (lane & 15)) * A_PITCH + (lane >> 4) * 8;
  union { v16bf v; v8bf h[2]; } u;
  u.h[0] = *(const v8bf*)(p);
  u.h[1] = *(const v8bf*)(p + 16);
  return u.v;
}

// B fragment, 32x16 bf16 (KxN), stored in LDS as [N][K] rows.
// Lane<16 holds N=lane, K={0..7,16..23}; lane>=16 holds N=lane-16, K={8..15,24..31}.
__device__ __forceinline__ v16bf frag_b(const __bf16* lds, int nBase, int lane) {
  const __bf16* p = lds + (nBase + (lane & 15)) * B_PITCH + (lane >> 4) * 8;
  union { v16bf v; v8bf h[2]; } u;
  u.h[0] = *(const v8bf*)(p);
  u.h[1] = *(const v8bf*)(p + 16);
  return u.v;
}

// Stage a [rows x 32] bf16 tile (row pitch = ld elems in global, A_PITCH in LDS).
// rows*2 16-byte chunks spread over 256 threads.
template <int ROWS>
__device__ __forceinline__ void load_tile(const __bf16* g, __bf16* lds, int ld,
                                          int k0, int tid) {
#pragma unroll
  for (int h = 0; h < (ROWS * 2) / 256; ++h) {
    const int c   = tid + h * 256;
    const int row = c >> 2;
    const int col = (c & 3) * 8;
    const __bf16* src = g + (size_t)row * ld + k0 + col;
    __bf16* dst = lds + row * A_PITCH + col;
#if USE_ASYNC_COPY
    async_ld_b128(src, dst);
#else
    *(v8bf*)dst = *(const v8bf*)src;
#endif
  }
}

// ---------------------------------------------------------------------------
// Kernel 1: QKV projection. Y = X @ W^T + b  (W is [out,in] row-major).
// z=0: Q = x1*Wq+bq -> row-major bf16 [B*SQ, DIM]
// z=1: K = x2*Wk+bk -> row-major bf16 [B*SK, DIM]
// z=2: V = x2*Wv+bv -> TRANSPOSED   bf16 Vt[b][d][k]  ([B, DIM, SK])
// f32 -> bf16 conversion must pass through VGPRs, so synchronous staging.
// ---------------------------------------------------------------------------
__global__ __launch_bounds__(256) void proj_kernel(
    const float* __restrict__ x1, const float* __restrict__ x2,
    const float* __restrict__ Wq, const float* __restrict__ bq,
    const float* __restrict__ Wk, const float* __restrict__ bk,
    const float* __restrict__ Wv, const float* __restrict__ bv,
    __bf16* __restrict__ Qo, __bf16* __restrict__ Ko, __bf16* __restrict__ Vt)
{
  const int z = blockIdx.z;
  const float* X    = (z == 0) ? x1 : x2;
  const float* W    = (z == 0) ? Wq : (z == 1) ? Wk : Wv;
  const float* bias = (z == 0) ? bq : (z == 1) ? bk : bv;

  __shared__ __bf16 ldsA[BLK_M * A_PITCH];
  __shared__ __bf16 ldsB[PRJ_N * B_PITCH];

  const int tid  = threadIdx.x;
  const int lane = tid & 31;
  const int wave = tid >> 5;
  const int wm   = wave & 3;   // 4 waves along M
  const int wn   = wave >> 2;  // 2 waves along N

  const int m0 = blockIdx.y * BLK_M;   // rows of X (0..8191)
  const int n0 = blockIdx.x * PRJ_N;   // output features (0..1023)

  const v8f zero = {0.f,0.f,0.f,0.f,0.f,0.f,0.f,0.f};
  v8f c[2][2] = {{zero, zero}, {zero, zero}};

  for (int k0 = 0; k0 < DIM; k0 += BLK_K) {
    // Stage A: 128x32 f32 -> bf16. 2 threads/row, 16 floats each.
    {
      const int row = tid >> 1;
      const int col = (tid & 1) * 16;
      const float* src = X + (size_t)(m0 + row) * DIM + k0 + col;
      v4f f0 = *(const v4f*)(src);
      v4f f1 = *(const v4f*)(src + 4);
      v4f f2 = *(const v4f*)(src + 8);
      v4f f3 = *(const v4f*)(src + 12);
      v8bf p0, p1;
#pragma unroll
      for (int i = 0; i < 4; ++i) {
        p0[i]     = (__bf16)f0[i];
        p0[i + 4] = (__bf16)f1[i];
        p1[i]     = (__bf16)f2[i];
        p1[i + 4] = (__bf16)f3[i];
      }
      __bf16* dst = ldsA + row * A_PITCH + col;
      *(v8bf*)(dst)     = p0;
      *(v8bf*)(dst + 8) = p1;
    }
    // Stage B: 64x32 f32 W rows -> bf16. 4 threads/row, 8 floats each.
    {
      const int row = tid >> 2;
      const int col = (tid & 3) * 8;
      const float* src = W + (size_t)(n0 + row) * DIM + k0 + col;
      v4f f0 = *(const v4f*)(src);
      v4f f1 = *(const v4f*)(src + 4);
      v8bf p0;
#pragma unroll
      for (int i = 0; i < 4; ++i) {
        p0[i]     = (__bf16)f0[i];
        p0[i + 4] = (__bf16)f1[i];
      }
      *(v8bf*)(ldsB + row * B_PITCH + col) = p0;
    }
    __syncthreads();

    const int mw = wm * 32;
    const int nw = wn * 32;
    v16bf a0 = frag_a(ldsA, mw,      lane);
    v16bf a1 = frag_a(ldsA, mw + 16, lane);
    v16bf b0 = frag_b(ldsB, nw,      lane);
    v16bf b1 = frag_b(ldsB, nw + 16, lane);
    c[0][0] = wmma_bf16f32(a0, b0, c[0][0]);
    c[0][1] = wmma_bf16f32(a0, b1, c[0][1]);
    c[1][0] = wmma_bf16f32(a1, b0, c[1][0]);
    c[1][1] = wmma_bf16f32(a1, b1, c[1][1]);
    __syncthreads();
  }

  // Epilogue: bias + convert + store.
#pragma unroll
  for (int i = 0; i < 2; ++i) {
#pragma unroll
    for (int j = 0; j < 2; ++j) {
      const int colg  = n0 + wn * 32 + j * 16 + (lane & 15);
      const int rbase = m0 + wm * 32 + i * 16 + (lane >> 4) * 8;
      const float bb  = bias[colg];
      v8f acc = c[i][j];
      if (z < 2) {
        __bf16* O = (z == 0) ? Qo : Ko;
#pragma unroll
        for (int r = 0; r < 8; ++r)
          O[(size_t)(rbase + r) * DIM + colg] = (__bf16)(acc[r] + bb);
      } else {
        // Transposed store: Vt[b][colg][s..s+7], contiguous 16B per lane.
        const int b  = rbase / SK;
        const int s0 = rbase % SK;
        v8bf pk;
#pragma unroll
        for (int r = 0; r < 8; ++r) pk[r] = (__bf16)(acc[r] + bb);
        *(v8bf*)(Vt + ((size_t)b * DIM + colg) * SK + s0) = pk;
      }
    }
  }
}

// ---------------------------------------------------------------------------
// Kernel 2/4: bf16 GEMM, C = A @ B^T (A and B stored [rows][K] row-major,
// leading dim = Kdim), f32 output. Batched via blockIdx.z.
// 128x128 block tile, double-buffered LDS, async global->LDS prefetch.
// 8 waves in 4(M) x 2(N); each wave: 32x64 = 2x4 WMMA tiles (8 WMMA / K-step).
// ---------------------------------------------------------------------------
__global__ __launch_bounds__(256) void gemm_bf16_kernel(
    const __bf16* __restrict__ Ag, const __bf16* __restrict__ Bg,
    float* __restrict__ Cg, int Kdim, int ldc,
    size_t strideA, size_t strideB, size_t strideC)
{
  __shared__ __bf16 ldsA[2][BLK_M * A_PITCH];
  __shared__ __bf16 ldsB[2][GEM_N * B_PITCH];

  const int tid  = threadIdx.x;
  const int lane = tid & 31;
  const int wave = tid >> 5;
  const int wm   = wave & 3;   // 4 waves along M
  const int wn   = wave >> 2;  // 2 waves along N

  const int m0 = blockIdx.y * BLK_M;
  const int n0 = blockIdx.x * GEM_N;

  const __bf16* Abase = Ag + (size_t)blockIdx.z * strideA + (size_t)m0 * Kdim;
  const __bf16* Bbase = Bg + (size_t)blockIdx.z * strideB + (size_t)n0 * Kdim;
  float* C = Cg + (size_t)blockIdx.z * strideC;

  const v8f zero = {0.f,0.f,0.f,0.f,0.f,0.f,0.f,0.f};
  v8f c[2][4];
#pragma unroll
  for (int i = 0; i < 2; ++i)
#pragma unroll
    for (int j = 0; j < 4; ++j) c[i][j] = zero;

  // Prologue: stage k0 = 0 into buffer 0.
  load_tile<BLK_M>(Abase, ldsA[0], Kdim, 0, tid);
  load_tile<GEM_N>(Bbase, ldsB[0], Kdim, 0, tid);

  int cur = 0;
  for (int k0 = 0; k0 < Kdim; k0 += BLK_K) {
    const bool haveNext = (k0 + BLK_K) < Kdim;
    if (haveNext) {
      // Prefetch next stage into the other buffer (overlaps with WMMA below).
      load_tile<BLK_M>(Abase, ldsA[cur ^ 1], Kdim, k0 + BLK_K, tid);
      load_tile<GEM_N>(Bbase, ldsB[cur ^ 1], Kdim, k0 + BLK_K, tid);
    }
#if USE_ASYNC_COPY
    // Async loads complete in order: <=4 outstanding (the 4 just issued)
    // guarantees this stage's 4 loads have landed in LDS.
    if (haveNext) __builtin_amdgcn_s_wait_asynccnt(4);
    else          __builtin_amdgcn_s_wait_asynccnt(0);
#endif
    __syncthreads();   // all waves' loads for 'cur' visible

    const __bf16* As = ldsA[cur];
    const __bf16* Bs = ldsB[cur];
    const int mw = wm * 32;
    const int nw = wn * 64;
    v16bf a0 = frag_a(As, mw,      lane);
    v16bf a1 = frag_a(As, mw + 16, lane);
    v16bf b0 = frag_b(Bs, nw,      lane);
    v16bf b1 = frag_b(Bs, nw + 16, lane);
    v16bf b2 = frag_b(Bs, nw + 32, lane);
    v16bf b3 = frag_b(Bs, nw + 48, lane);
    c[0][0] = wmma_bf16f32(a0, b0, c[0][0]);
    c[0][1] = wmma_bf16f32(a0, b1, c[0][1]);
    c[0][2] = wmma_bf16f32(a0, b2, c[0][2]);
    c[0][3] = wmma_bf16f32(a0, b3, c[0][3]);
    c[1][0] = wmma_bf16f32(a1, b0, c[1][0]);
    c[1][1] = wmma_bf16f32(a1, b1, c[1][1]);
    c[1][2] = wmma_bf16f32(a1, b2, c[1][2]);
    c[1][3] = wmma_bf16f32(a1, b3, c[1][3]);
    __syncthreads();   // all waves done reading 'cur' -> safe to refill it
    cur ^= 1;
  }

#pragma unroll
  for (int i = 0; i < 2; ++i) {
#pragma unroll
    for (int j = 0; j < 4; ++j) {
      const int colg  = n0 + wn * 64 + j * 16 + (lane & 15);
      const int rbase = m0 + wm * 32 + i * 16 + (lane >> 4) * 8;
      v8f acc = c[i][j];
#pragma unroll
      for (int r = 0; r < 8; ++r)
        C[(size_t)(rbase + r) * ldc + colg] = acc[r];
    }
  }
}

// ---------------------------------------------------------------------------
// Kernel 3: row softmax with scale 1/sqrt(DIM)=1/32. One block per row.
// Reads f32 scores, writes bf16 probabilities.
// ---------------------------------------------------------------------------
__global__ __launch_bounds__(256) void softmax_kernel(
    const float* __restrict__ S, __bf16* __restrict__ P)
{
  const size_t row = blockIdx.x;
  const float* s = S + row * (size_t)SK;
  __bf16* p = P + row * (size_t)SK;
  const int tid = threadIdx.x;

  float v[8];
  float m = -3.0e38f;
#pragma unroll
  for (int i = 0; i < 8; ++i) {
    v[i] = s[tid + i * 256] * 0.03125f;   // scale = 1/sqrt(1024)
    m = fmaxf(m, v[i]);
  }

  __shared__ float red[256];
  red[tid] = m;
  __syncthreads();
  for (int off = 128; off > 0; off >>= 1) {
    if (tid < off) red[tid] = fmaxf(red[tid], red[tid + off]);
    __syncthreads();
  }
  m = red[0];
  __syncthreads();

  float sum = 0.f;
#pragma unroll
  for (int i = 0; i < 8; ++i) {
    v[i] = __expf(v[i] - m);
    sum += v[i];
  }
  red[tid] = sum;
  __syncthreads();
  for (int off = 128; off > 0; off >>= 1) {
    if (tid < off) red[tid] += red[tid + off];
    __syncthreads();
  }
  const float inv = 1.0f / red[0];
#pragma unroll
  for (int i = 0; i < 8; ++i)
    p[tid + i * 256] = (__bf16)(v[i] * inv);
}

// ---------------------------------------------------------------------------
extern "C" void kernel_launch(void* const* d_in, const int* in_sizes, int n_in,
                              void* d_out, int out_size, void* d_ws, size_t ws_size,
                              hipStream_t stream)
{
  (void)in_sizes; (void)n_in; (void)out_size; (void)ws_size;
  const float* x1 = (const float*)d_in[0];
  const float* x2 = (const float*)d_in[1];
  const float* Wq = (const float*)d_in[2];
  const float* bq = (const float*)d_in[3];
  const float* Wk = (const float*)d_in[4];
  const float* bk = (const float*)d_in[5];
  const float* Wv = (const float*)d_in[6];
  const float* bv = (const float*)d_in[7];
  float* out = (float*)d_out;

  // Workspace layout (bytes):
  const size_t qsz = (size_t)NBATCH * SQ * DIM * 2;   // 16 MB each
  const size_t ssz = (size_t)NBATCH * SQ * SK * 4;    // 67 MB
  char* ws = (char*)d_ws;
  __bf16* Qb = (__bf16*)(ws);
  __bf16* Kb = (__bf16*)(ws + qsz);
  __bf16* Vt = (__bf16*)(ws + 2 * qsz);
  float*  Sm = (float*)(ws + 3 * qsz);
  __bf16* Pm = (__bf16*)(ws + 3 * qsz + ssz);

  const dim3 blk(256);

  // 1) QKV projections (z: 0=Q, 1=K, 2=V-transposed)
  proj_kernel<<<dim3(DIM / PRJ_N, (NBATCH * SQ) / BLK_M, 3), blk, 0, stream>>>(
      x1, x2, Wq, bq, Wk, bk, Wv, bv, Qb, Kb, Vt);

  // 2) S = Q @ K^T  (per batch; A ld = DIM, B ld = DIM, C ld = SK)
  gemm_bf16_kernel<<<dim3(SK / GEM_N, SQ / BLK_M, NBATCH), blk, 0, stream>>>(
      Qb, Kb, Sm, DIM, SK,
      (size_t)SQ * DIM, (size_t)SK * DIM, (size_t)SQ * SK);

  // 3) P = softmax(S * 1/32) row-wise, bf16 out
  softmax_kernel<<<dim3(NBATCH * SQ), blk, 0, stream>>>(Sm, Pm);

  // 4) O = P @ V = P @ Vt^T  (A ld = SK, B ld = SK, C ld = DIM)
  gemm_bf16_kernel<<<dim3(DIM / GEM_N, SQ / BLK_M, NBATCH), blk, 0, stream>>>(
      Pm, Vt, out, SK, DIM,
      (size_t)SQ * SK, (size_t)DIM * SK, (size_t)SQ * DIM);
}
